// TransformerDeter_9869834847026
// MI455X (gfx1250) — matrix operand order repaired and checked by reading, not compiled, not run on hardware
//
#include <hip/hip_runtime.h>

typedef unsigned short u16;
typedef __attribute__((ext_vector_type(16))) __bf16 v16bf;
typedef __attribute__((ext_vector_type(8)))  float  v8f;
typedef __attribute__((ext_vector_type(8)))  unsigned short u16x8;
typedef __attribute__((ext_vector_type(4)))  unsigned int u32x4;
typedef __attribute__((ext_vector_type(8)))  int i32x8;
typedef __attribute__((ext_vector_type(4)))  int i32x4;

#define DEV __device__ __forceinline__

DEV float bf2f(u16 u) { union { unsigned i; float f; } x; x.i = ((unsigned)u) << 16; return x.f; }
DEV u16 f2bf(float f) {
  union { unsigned i; float f; } x; x.f = f;
  unsigned i = x.i;
  return (u16)((i + 0x7FFFu + ((i >> 16) & 1u)) >> 16);  // RNE
}

union FragU { v16bf v; float4 f[2]; };

// A-fragment (16x32 bf16): lane<16 -> M=lane, K {0..7, 16..23}; lane>=16 -> K {8..15, 24..31}
DEV v16bf fragA(const u16* base, int stride, int row, int k0, int lane) {
  const u16* p = base + (row + (lane & 15)) * stride + k0 + ((lane & 16) ? 8 : 0);
  FragU f; f.f[0] = *(const float4*)p; f.f[1] = *(const float4*)(p + 16); return f.v;
}
// B-fragment (32x16 bf16) from LDS stored [n][k]: lane<16 -> N=lane, K {0..7, 16..23}; lane>=16 -> K {8..15, 24..31}
DEV v16bf fragB(const u16* base, int stride, int col, int k0, int lane) {
  const u16* p = base + (col + (lane & 15)) * stride + k0 + ((lane & 16) ? 8 : 0);
  FragU f; f.f[0] = *(const float4*)p; f.f[1] = *(const float4*)(p + 16); return f.v;
}
DEV v8f wmma_bf(v16bf a, v16bf b, v8f c) {
  return __builtin_amdgcn_wmma_f32_16x16x32_bf16(false, a, false, b, (short)0, c, false, false);
}
DEV v8f vzero8() { v8f z; for (int i = 0; i < 8; ++i) z[i] = 0.f; return z; }

DEV unsigned lds_off(const void* p) { return (unsigned)(unsigned long long)p; }

// ---------------------------------------------------------------------------
// TDM: DMA a 2D tile (tile_dim0 = 32 bf16 elems per row, tile_dim1 = 128 rows)
// from a row-major tensor with row stride `stride_elems` into LDS at lds_addr,
// padding each 64B row with 16B so the LDS layout has stride 40 u16 (matches
// fragment loaders). Rows >= tensor_rows are zero-filled (OOB reads -> 0).
// D# layout per CDNA5 ISA ch.8 (group0 128b, group1 256b; groups 2/3 zero).
// Toolchain here is the 6-arg builtin: (u32x4, i32x8, i32x4, i32x4, i32x8, i32)
// ---------------------------------------------------------------------------
DEV void tdm_load_tile(const u16* gp, unsigned ldsaddr, unsigned tensor_rows,
                       unsigned long long stride_elems) {
  unsigned long long ga = (unsigned long long)(const void*)gp;
  u32x4 g0;
  g0[0] = 1u;                                              // count=1 (valid user D#)
  g0[1] = ldsaddr;                                         // lds_addr
  g0[2] = (unsigned)ga;                                    // global_addr[31:0]
  g0[3] = (unsigned)((ga >> 32) & 0x1FFFFFFull) | (2u << 30); // addr[56:32] | type=2
  i32x8 g1;
  // data_size=1 (2B), pad_enable, pad_interval=3 (16 DW=64B), pad_amount=3 (4 DW=16B)
  g1[0] = (int)((1u << 16) | (1u << 20) | (3u << 22) | (3u << 25));
  g1[1] = (int)(32u << 16);                                // tensor_dim0 = 32
  g1[2] = (int)((tensor_rows & 0xFFFFu) << 16);            // tensor_dim1 lo
  g1[3] = (int)((tensor_rows >> 16) | (32u << 16));        // tensor_dim1 hi | tile_dim0=32
  g1[4] = (int)128;                                        // tile_dim1 = 128 rows
  g1[5] = (int)(unsigned)(stride_elems & 0xFFFFFFFFull);   // tensor_dim0_stride lo
  g1[6] = (int)(unsigned)((stride_elems >> 32) & 0xFFFFull);
  g1[7] = 0;
  i32x4 z4; z4[0] = z4[1] = z4[2] = z4[3] = 0;
  i32x8 z8; for (int i = 0; i < 8; ++i) z8[i] = 0;
  __builtin_amdgcn_tensor_load_to_lds(g0, g1, z4, z4, z8, 0);
}

// ---------------------------------------------------------------------------
// Generic bf16 GEMM: C(MxN) = A(MxK) x B^T where Bw is bf16 [N][K] row-major
// (original weight layout). TDM double-buffered LDS staging, 8 waves, 128x128
// block tile, each wave a 32x64 C tile via v_wmma_f32_16x16x32_bf16.
// EPI: 0 = store bf16 (no bias); 2 = bias+GELU -> bf16;
//      3 = bias + residual(f32) -> f32; 4 = bias -> f32
// ---------------------------------------------------------------------------
template <int EPI>
__global__ __launch_bounds__(256) void gemm_bf16_kernel(
    const u16* __restrict__ A, const u16* __restrict__ Bw,
    int M, int N, int K,
    const float* __restrict__ bias, const float* __restrict__ resid,
    u16* __restrict__ outb, float* __restrict__ outf)
{
  __shared__ __align__(16) u16 As[2][128 * 40];   // [m][k], stride 40
  __shared__ __align__(16) u16 Bs[2][128 * 40];   // [n][k], stride 40
  const int tid = threadIdx.x;
  const int lane = tid & 31, wave = tid >> 5;
  const int wm = wave >> 1, wn = wave & 1;               // 4 x 2 wave grid
  const int m0 = blockIdx.y * 128, n0 = blockIdx.x * 128;
  const int nk = K >> 5;
  const unsigned mrows = (M - m0 < 128) ? (unsigned)(M - m0) : 128u;

  v8f acc[2][4];
  for (int mt = 0; mt < 2; ++mt) for (int nt = 0; nt < 4; ++nt) acc[mt][nt] = vzero8();

  if (wave == 0) {
    tdm_load_tile(A  + (size_t)m0 * K, lds_off(&As[0][0]), mrows, (unsigned long long)K);
    tdm_load_tile(Bw + (size_t)n0 * K, lds_off(&Bs[0][0]), 128u,  (unsigned long long)K);
  }
  for (int it = 0; it < nk; ++it) {
    if (wave == 0) {
      if (it + 1 < nk) {
        int kb2 = (it + 1) << 5;
        tdm_load_tile(A  + (size_t)m0 * K + kb2, lds_off(&As[(it + 1) & 1][0]), mrows, (unsigned long long)K);
        tdm_load_tile(Bw + (size_t)n0 * K + kb2, lds_off(&Bs[(it + 1) & 1][0]), 128u,  (unsigned long long)K);
        __builtin_amdgcn_s_wait_tensorcnt((short)2);   // current tile complete
      } else {
        __builtin_amdgcn_s_wait_tensorcnt((short)0);
      }
    }
    __syncthreads();
    const u16* as = &As[it & 1][0];
    const u16* bs = &Bs[it & 1][0];
    v16bf af0 = fragA(as, 40, wm * 32,      0, lane);
    v16bf af1 = fragA(as, 40, wm * 32 + 16, 0, lane);
    for (int nt = 0; nt < 4; ++nt) {
      v16bf bfr = fragB(bs, 40, wn * 64 + nt * 16, 0, lane);
      acc[0][nt] = wmma_bf(af0, bfr, acc[0][nt]);
      acc[1][nt] = wmma_bf(af1, bfr, acc[1][nt]);
    }
    __syncthreads();
  }
  for (int mt = 0; mt < 2; ++mt)
    for (int nt = 0; nt < 4; ++nt)
      for (int vg = 0; vg < 8; ++vg) {
        int m = m0 + wm * 32 + mt * 16 + vg + ((lane & 16) ? 8 : 0);
        int n = n0 + wn * 64 + nt * 16 + (lane & 15);
        if (m >= M) continue;
        float val = acc[mt][nt][vg];
        if (EPI != 0) val += bias[n];
        if (EPI == 2) val = 0.5f * val * (1.f + erff(val * 0.70710678118f));
        size_t idx = (size_t)m * N + n;
        if (EPI == 3)      outf[idx] = resid[idx] + val;
        else if (EPI == 4) outf[idx] = val;
        else               outb[idx] = f2bf(val);
      }
}

// ---------------------------------------------------------------------------
// Fused rel-attention per (b,h): AC/BD WMMA -> rel_shift+mask+softmax -> P@V
// qkv layout: [b][t][3][H=16][DH=64] bf16; rhg: [t][1024] bf16; ctx: [b*64+t][1024]
// ---------------------------------------------------------------------------
__global__ __launch_bounds__(128) void attn_kernel(
    const u16* __restrict__ qkv, const u16* __restrict__ rhg,
    const float* __restrict__ uu, const float* __restrict__ vv,
    u16* __restrict__ ctx)
{
  __shared__ __align__(16) char smem[63488];
  u16*   quP  = (u16*)smem;               // [64][72]; reused for probs later
  u16*   qvP  = (u16*)(smem + 9216);      // [64][72]
  u16*   kkP  = (u16*)(smem + 18432);     // [64][72]
  u16*   rhP  = (u16*)(smem + 27648);     // [64][72]
  u16*   vtP  = (u16*)(smem + 36864);     // [64][72]  V transposed [d][s]
  float* acP  = (float*)(smem + 9216);    // [64][64]  aliases qv/kk (dead)
  float* bdP  = (float*)(smem + 46080);   // [64][64]
  float* redP = (float*)(smem + 62464);   // [64][4]
  u16*   prP  = quP;

  const int bh = blockIdx.x;
  const int b = bh >> 4, h = bh & 15;
  const int tid = threadIdx.x;
  const int lane = tid & 31, wave = tid >> 5;

  for (int i = tid; i < 64 * 8; i += 128) {
    int t = i >> 3, c8 = (i & 7) * 8;
    size_t base = ((size_t)(b * 64 + t)) * 3072 + h * 64 + c8;
    u16x8 q8 = *(const u16x8*)(qkv + base);
    u16x8 k8 = *(const u16x8*)(qkv + base + 1024);
    u16x8 v8 = *(const u16x8*)(qkv + base + 2048);
    u16x8 r8 = *(const u16x8*)(rhg + t * 1024 + h * 64 + c8);
    for (int j = 0; j < 8; ++j) {
      int d = c8 + j;
      float qf = bf2f(q8[j]);
      quP[t * 72 + d] = f2bf(qf + uu[h * 64 + d]);
      qvP[t * 72 + d] = f2bf(qf + vv[h * 64 + d]);
      kkP[t * 72 + d] = k8[j];
      vtP[d * 72 + t] = v8[j];
      rhP[t * 72 + d] = r8[j];
    }
  }
  __syncthreads();

  v8f ac[4], bd[4];
  for (int nt = 0; nt < 4; ++nt) { ac[nt] = vzero8(); bd[nt] = vzero8(); }
  const int m0 = wave * 16;
  for (int k0 = 0; k0 < 64; k0 += 32) {
    v16bf au = fragA(quP, 72, m0, k0, lane);
    v16bf av = fragA(qvP, 72, m0, k0, lane);
    for (int nt = 0; nt < 4; ++nt) {
      v16bf bk = fragB(kkP, 72, nt * 16, k0, lane);
      v16bf br = fragB(rhP, 72, nt * 16, k0, lane);
      ac[nt] = wmma_bf(au, bk, ac[nt]);
      bd[nt] = wmma_bf(av, br, bd[nt]);
    }
  }
  __syncthreads();  // all waves done reading tiles before aliasing writes
  for (int nt = 0; nt < 4; ++nt)
    for (int vg = 0; vg < 8; ++vg) {
      int t = m0 + vg + ((lane & 16) ? 8 : 0);
      int s = nt * 16 + (lane & 15);
      acP[t * 64 + s] = ac[nt][vg];
      bdP[t * 64 + s] = bd[nt][vg];
    }
  __syncthreads();

  {
    const int t = tid >> 1, half = tid & 1;
    float sc[32];
    float mx = -1e30f;
    for (int i = 0; i < 32; ++i) {
      int s = half * 32 + i;
      float val = -1e30f;
      if (s <= t) val = (acP[t * 64 + s] + bdP[t * 64 + (s - t + 63)]) * 0.125f;
      sc[i] = val; mx = fmaxf(mx, val);
    }
    redP[t * 4 + half] = mx;
    __syncthreads();
    mx = fmaxf(redP[t * 4], redP[t * 4 + 1]);
    float sum = 0.f;
    for (int i = 0; i < 32; ++i) { sc[i] = expf(sc[i] - mx); sum += sc[i]; }
    redP[t * 4 + 2 + half] = sum;
    __syncthreads();
    float inv = 1.f / (redP[t * 4 + 2] + redP[t * 4 + 3]);
    for (int i = 0; i < 32; ++i)
      prP[t * 72 + half * 32 + i] = f2bf(sc[i] * inv);
  }
  __syncthreads();

  v8f ov[4];
  for (int nt = 0; nt < 4; ++nt) ov[nt] = vzero8();
  for (int k0 = 0; k0 < 64; k0 += 32) {
    v16bf a = fragA(prP, 72, m0, k0, lane);
    for (int nt = 0; nt < 4; ++nt) {
      v16bf bv = fragB(vtP, 72, nt * 16, k0, lane);
      ov[nt] = wmma_bf(a, bv, ov[nt]);
    }
  }
  for (int nt = 0; nt < 4; ++nt)
    for (int vg = 0; vg < 8; ++vg) {
      int t = m0 + vg + ((lane & 16) ? 8 : 0);
      int d = nt * 16 + (lane & 15);
      ctx[((size_t)(b * 64 + t)) * 1024 + h * 64 + d] = f2bf(ov[nt][vg]);
    }
}

// ---------------------------------------------------------------------------
// Elementwise / norm kernels
// ---------------------------------------------------------------------------
__global__ __launch_bounds__(256) void rmsnorm_cast_kernel(
    const float* __restrict__ xin, const float* __restrict__ wgt, u16* __restrict__ out)
{
  const int row = blockIdx.x, tid = threadIdx.x;
  const float* xr = xin + (size_t)row * 1024;
  float4 v = *(const float4*)(xr + tid * 4);
  __shared__ float red[256];
  red[tid] = v.x * v.x + v.y * v.y + v.z * v.z + v.w * v.w;
  __syncthreads();
  for (int off = 128; off > 0; off >>= 1) { if (tid < off) red[tid] += red[tid + off]; __syncthreads(); }
  float sc = rsqrtf(red[0] * (1.f / 1024.f) + 1e-4f);
  u16* o = out + (size_t)row * 1024 + tid * 4;
  const float* wp = wgt + tid * 4;
  o[0] = f2bf(v.x * sc * wp[0]); o[1] = f2bf(v.y * sc * wp[1]);
  o[2] = f2bf(v.z * sc * wp[2]); o[3] = f2bf(v.w * sc * wp[3]);
}

__global__ __launch_bounds__(256) void token_kernel(
    const float* __restrict__ t0, const float* __restrict__ wgt,
    float* __restrict__ x, float* __restrict__ omem)
{
  const int b = blockIdx.x, tid = threadIdx.x;
  float4 v = *(const float4*)(t0 + (size_t)b * 1024 + tid * 4);
  __shared__ float red[256];
  red[tid] = v.x * v.x + v.y * v.y + v.z * v.z + v.w * v.w;
  __syncthreads();
  for (int off = 128; off > 0; off >>= 1) { if (tid < off) red[tid] += red[tid + off]; __syncthreads(); }
  float sc = rsqrtf(red[0] * (1.f / 1024.f) + 1e-4f);
  float vv[4] = {v.x, v.y, v.z, v.w};
  size_t base = ((size_t)(b * 64 + 63)) * 1024 + tid * 4;
  for (int j = 0; j < 4; ++j) {
    float y = vv[j] * sc * wgt[tid * 4 + j];
    y = y / (1.f + expf(-y));               // silu
    x[base + j] = y; omem[base + j] = y;
  }
}

__global__ __launch_bounds__(256) void finalnorm_kernel(
    const float* __restrict__ x, const float* __restrict__ wgt, float* __restrict__ outd)
{
  const int b = blockIdx.x, tid = threadIdx.x;
  float4 v = *(const float4*)(x + ((size_t)(b * 64 + 63)) * 1024 + tid * 4);
  __shared__ float red[256];
  red[tid] = v.x * v.x + v.y * v.y + v.z * v.z + v.w * v.w;
  __syncthreads();
  for (int off = 128; off > 0; off >>= 1) { if (tid < off) red[tid] += red[tid + off]; __syncthreads(); }
  float sc = rsqrtf(red[0] * (1.f / 1024.f) + 1e-4f);
  float vv[4] = {v.x, v.y, v.z, v.w};
  for (int j = 0; j < 4; ++j)
    outd[(size_t)b * 1024 + tid * 4 + j] = vv[j] * sc * wgt[tid * 4 + j];
}

__global__ __launch_bounds__(256) void prep_inp_kernel(
    const float* __restrict__ stoch, const float* __restrict__ action, u16* __restrict__ inp)
{
  int i = blockIdx.x * 256 + threadIdx.x;
  if (i >= 128 * 1280) return;
  int b = i / 1280, c = i % 1280;
  float v;
  if (c < 1024) v = stoch[b * 1024 + c];
  else { float a = action[b * 256 + (c - 1024)]; v = a / fmaxf(fabsf(a), 1.0f); }
  inp[i] = f2bf(v);
}

__global__ __launch_bounds__(256) void shift_mem_kernel(
    const float* __restrict__ mem, float* __restrict__ x, float* __restrict__ omem)
{
  int i = blockIdx.x * 256 + threadIdx.x;
  if (i >= 128 * 63 * 1024) return;
  int b = i / (63 * 1024), rem = i % (63 * 1024);
  int t = rem >> 10, d = rem & 1023;
  float v = mem[((size_t)(b * 64 + t + 1)) * 1024 + d];
  size_t o = ((size_t)(b * 64 + t)) * 1024 + d;
  x[o] = v; omem[o] = v;
}

__global__ __launch_bounds__(256) void relpos_kernel(u16* __restrict__ r)
{
  const int t = blockIdx.x, tid = threadIdx.x;
  float pos = (float)(63 - t);
  for (int j = tid; j < 512; j += 256) {
    float freq = expf(-((float)(2 * j) / 1024.f) * 9.210340371976184f);  // 10000^(-2j/1024)
    float a = pos * freq;
    r[t * 1024 + j]       = f2bf(sinf(a));
    r[t * 1024 + 512 + j] = f2bf(cosf(a));
  }
}

__global__ __launch_bounds__(256) void cast_bf16_kernel(
    const float* __restrict__ W, u16* __restrict__ O, int n8)
{
  int i = blockIdx.x * 256 + threadIdx.x;
  if (i >= n8) return;
  const float4* p = (const float4*)(W + (size_t)i * 8);
  float4 a = p[0], b = p[1];
  u16x8 o;
  o[0] = f2bf(a.x); o[1] = f2bf(a.y); o[2] = f2bf(a.z); o[3] = f2bf(a.w);
  o[4] = f2bf(b.x); o[5] = f2bf(b.y); o[6] = f2bf(b.z); o[7] = f2bf(b.w);
  *(u16x8*)(O + (size_t)i * 8) = o;
}

// ---------------------------------------------------------------------------
extern "C" void kernel_launch(void* const* d_in, const int* in_sizes, int n_in,
                              void* d_out, int out_size, void* d_ws, size_t ws_size,
                              hipStream_t stream) {
  (void)in_sizes; (void)n_in; (void)out_size; (void)ws_size;
  const float* stoch      = (const float*)d_in[0];
  const float* memory     = (const float*)d_in[2];
  const float* action     = (const float*)d_in[3];
  const float* in_w       = (const float*)d_in[4];
  const float* in_b       = (const float*)d_in[5];
  const float* in_norm_w  = (const float*)d_in[6];
  const float* norm1_w    = (const float*)d_in[7];
  const float* qkv_w      = (const float*)d_in[8];
  const float* r_w        = (const float*)d_in[9];
  const float* u_w        = (const float*)d_in[10];
  const float* v_w        = (const float*)d_in[11];
  const float* attn_out_w = (const float*)d_in[12];
  const float* attn_out_b = (const float*)d_in[13];
  const float* norm2_w    = (const float*)d_in[14];
  const float* ff1_w      = (const float*)d_in[15];
  const float* ff1_b      = (const float*)d_in[16];
  const float* ff2_w      = (const float*)d_in[17];
  const float* ff2_b      = (const float*)d_in[18];
  const float* out_norm_w = (const float*)d_in[19];

  float* out_deter = (float*)d_out;
  float* out_mem   = (float*)d_out + (size_t)128 * 1024;

  char* wsp = (char*)d_ws;
  auto alloc = [&](size_t bytes) { char* p = wsp; wsp += (bytes + 255) & ~(size_t)255; return p; };
  u16*   inp   = (u16*)alloc((size_t)128 * 1280 * 2);
  u16*   in_wB = (u16*)alloc((size_t)1024 * 1280 * 2);          // [1024][1280] bf16
  u16*   qkvB  = (u16*)alloc((size_t)4 * 3072 * 1024 * 2);      // [3072][1024] per layer
  u16*   rB    = (u16*)alloc((size_t)4 * 1024 * 1024 * 2);
  u16*   outB  = (u16*)alloc((size_t)4 * 1024 * 1024 * 2);
  u16*   ff1B  = (u16*)alloc((size_t)4 * 4096 * 1024 * 2);
  u16*   ff2B  = (u16*)alloc((size_t)4 * 1024 * 4096 * 2);
  float* x     = (float*)alloc((size_t)8192 * 1024 * 4);
  u16*   xn    = (u16*)alloc((size_t)8192 * 1024 * 2);
  u16*   qkv   = (u16*)alloc((size_t)8192 * 3072 * 2);
  u16*   rpe   = (u16*)alloc((size_t)64 * 1024 * 2);
  u16*   rh    = (u16*)alloc((size_t)64 * 1024 * 2);
  u16*   ctx   = (u16*)alloc((size_t)8192 * 1024 * 2);
  u16*   hff   = (u16*)alloc((size_t)8192 * 4096 * 2);
  float* t0    = (float*)alloc((size_t)128 * 1024 * 4);

  auto castw = [&](const float* W, u16* O, size_t n) {
    int n8 = (int)(n / 8);
    cast_bf16_kernel<<<(n8 + 255) / 256, 256, 0, stream>>>(W, O, n8);
  };
  castw(in_w,       in_wB, (size_t)1024 * 1280);
  castw(qkv_w,      qkvB,  (size_t)4 * 3072 * 1024);
  castw(r_w,        rB,    (size_t)4 * 1024 * 1024);
  castw(attn_out_w, outB,  (size_t)4 * 1024 * 1024);
  castw(ff1_w,      ff1B,  (size_t)4 * 4096 * 1024);
  castw(ff2_w,      ff2B,  (size_t)4 * 1024 * 4096);

  prep_inp_kernel<<<(128 * 1280 + 255) / 256, 256, 0, stream>>>(stoch, action, inp);
  // new_token = silu(rmsnorm(inp @ in_w.T + in_b))
  gemm_bf16_kernel<4><<<dim3(1024 / 128, 1), 256, 0, stream>>>(
      inp, in_wB, 128, 1024, 1280, in_b, nullptr, nullptr, t0);
  token_kernel<<<128, 256, 0, stream>>>(t0, in_norm_w, x, out_mem);
  shift_mem_kernel<<<(128 * 63 * 1024 + 255) / 256, 256, 0, stream>>>(memory, x, out_mem);
  relpos_kernel<<<64, 256, 0, stream>>>(rpe);

  for (int l = 0; l < 4; ++l) {
    rmsnorm_cast_kernel<<<8192, 256, 0, stream>>>(x, norm1_w + (size_t)l * 1024, xn);
    gemm_bf16_kernel<0><<<dim3(3072 / 128, 8192 / 128), 256, 0, stream>>>(
        xn, qkvB + (size_t)l * 3072 * 1024, 8192, 3072, 1024, nullptr, nullptr, qkv, nullptr);
    gemm_bf16_kernel<0><<<dim3(1024 / 128, 1), 256, 0, stream>>>(
        rpe, rB + (size_t)l * 1024 * 1024, 64, 1024, 1024, nullptr, nullptr, rh, nullptr);
    attn_kernel<<<128 * 16, 128, 0, stream>>>(
        qkv, rh, u_w + (size_t)l * 1024, v_w + (size_t)l * 1024, ctx);
    gemm_bf16_kernel<3><<<dim3(1024 / 128, 8192 / 128), 256, 0, stream>>>(
        ctx, outB + (size_t)l * 1024 * 1024, 8192, 1024, 1024,
        attn_out_b + (size_t)l * 1024, x, nullptr, x);
    rmsnorm_cast_kernel<<<8192, 256, 0, stream>>>(x, norm2_w + (size_t)l * 1024, xn);
    gemm_bf16_kernel<2><<<dim3(4096 / 128, 8192 / 128), 256, 0, stream>>>(
        xn, ff1B + (size_t)l * 4096 * 1024, 8192, 4096, 1024,
        ff1_b + (size_t)l * 4096, nullptr, hff, nullptr);
    gemm_bf16_kernel<3><<<dim3(1024 / 128, 8192 / 128), 256, 0, stream>>>(
        hff, ff2B + (size_t)l * 4096 * 1024, 8192, 1024, 4096,
        ff2_b + (size_t)l * 1024, x, nullptr, x);
  }
  finalnorm_kernel<<<128, 256, 0, stream>>>(x, out_norm_w, out_deter);
}
